// LinearLayerDecoder_78116865180357
// MI455X (gfx1250) — matrix operand order repaired and checked by reading, not compiled, not run on hardware
//
#include <hip/hip_runtime.h>
#include <hip/hip_bf16.h>

// ---------------------------------------------------------------------------
// CDNA5 / gfx1250 edge-MLP decoder.
//   P = [ z@W1[0:128] + b1 | z@W1[128:256] ]  (precompute, bf16, in d_ws)
//   pack: W2,W3 -> bf16 transposed+padded images, W4 -> 16-row image (d_ws)
//   per edge: h1 = relu(P_a[src] + P_b[dst] + g*W1[256])    (f32 VALU, bf16 I/O)
//             h2 = relu(h1 @ W2 + b2)   -> v_wmma_f32_16x16x32_bf16
//             h3 = relu(h2 @ W3 + b3)   -> v_wmma_f32_16x16x32_bf16
//             val = h3 @ W4 + b4        -> v_wmma (W4 image, row 0)
//             out = sigmoid(val)
// Weight images staged into LDS via TENSOR_LOAD_TO_LDS (TDM) + TENSORcnt wait.
// ---------------------------------------------------------------------------

typedef __attribute__((ext_vector_type(16))) __bf16 v16bf;
typedef __attribute__((ext_vector_type(8)))  __bf16 v8bf;
typedef __attribute__((ext_vector_type(4)))  __bf16 v4bf;
typedef __attribute__((ext_vector_type(8)))  float  v8f;
typedef __attribute__((ext_vector_type(4)))  float  v4f;

typedef unsigned int uint32x4 __attribute__((ext_vector_type(4)));
typedef int          int32x8  __attribute__((ext_vector_type(8)));
typedef int          int32x4  __attribute__((ext_vector_type(4)));

#define WAVES_PER_WG 4
#define TILE_M       32    // edges per wave (two 16-wide WMMA N-tiles)
#define KPAD         136   // padded row length (bf16 elems); 272B rows, 16B aligned
#define WBYTES       (128 * KPAD * 2)   // one packed weight image (34816 B)
#define W4ELEMS      (16 * KPAD)        // W4 image elements (row 0 = W4)

__device__ __forceinline__ void lds_wave_sync() {
  asm volatile("s_wait_dscnt 0" ::: "memory");
  __builtin_amdgcn_wave_barrier();
}

__device__ __forceinline__ v16bf cat8(v8bf lo, v8bf hi) {
  v16bf r;
#pragma unroll
  for (int i = 0; i < 8; ++i) { r[i] = lo[i]; r[i + 8] = hi[i]; }
  return r;
}

// f32 relu (dual-issued v_max pairs) then paired v_cvt_pk_bf16_f32.
__device__ __forceinline__ v8bf relu_cvt8(v8f a) {
  v8f r;
#pragma unroll
  for (int i = 0; i < 8; ++i)
    r[i] = __builtin_amdgcn_fmed3f(a[i], 0.0f, __builtin_inff());
  return __builtin_convertvector(r, v8bf);
}

// Issue one TDM transfer: 1 x nelem (2-byte elems) global image -> LDS.
__device__ __forceinline__ void tdm_load_to_lds(const void* gsrc, unsigned lds_off,
                                                unsigned nelem) {
  const unsigned long long ga = (unsigned long long)gsrc;
  uint32x4 g0;
  g0[0] = 1u;                                          // count=1 (valid), user mode
  g0[1] = lds_off;                                     // lds_addr (bytes)
  g0[2] = (unsigned)(ga & 0xFFFFFFFFu);                // global_addr[31:0]
  g0[3] = (unsigned)((ga >> 32) & 0x01FFFFFFu)         // global_addr[56:32]
          | 0x80000000u;                               // type = 2 ("image")
  int32x8 g1;
  g1[0] = (int)(1u << 16);                             // data_size=1 (2B), mask=0
  g1[1] = (int)((nelem & 0xFFFFu) << 16);              // tensor_dim0[15:0]
  g1[2] = (int)(((nelem >> 16) & 0xFFFFu) | (1u << 16)); // dim0[31:16] | tensor_dim1=1
  g1[3] = (int)((nelem & 0xFFFFu) << 16);              // tile_dim0
  g1[4] = 1;                                           // tile_dim1 = 1
  g1[5] = (int)nelem;                                  // tensor_dim0_stride[31:0]
  g1[6] = 0;
  g1[7] = 0;
  const int32x4 z4 = {0, 0, 0, 0};
  const int32x8 z8 = {0, 0, 0, 0, 0, 0, 0, 0};
  __builtin_amdgcn_tensor_load_to_lds(g0, g1, z4, z4, z8, 0);
}

// ---------------------------------------------------------------------------
// Precompute P (bf16): P[n][0:128] = z[n]@W1[0:128,:] + b1 ; [128:256] = z[n]@W1[128:256,:]
// ---------------------------------------------------------------------------
__global__ __launch_bounds__(256) void precompute_P_kernel(
    const float* __restrict__ z, const float* __restrict__ W1,
    const float* __restrict__ b1, __bf16* __restrict__ P, int n_nodes)
{
  __shared__ float zrow[128];
  const int n = blockIdx.x;
  const int j = threadIdx.x;
  if (j < 128) zrow[j] = z[(long long)n * 128 + j];
  __syncthreads();
  const float* wbase = W1 + ((j < 128) ? j : (128 * 128 + (j - 128)));
  float acc = (j < 128) ? b1[j] : 0.0f;
#pragma unroll 8
  for (int k = 0; k < 128; ++k)
    acc = fmaf(zrow[k], wbase[(long long)k * 128], acc);
  P[(long long)n * 256 + j] = (__bf16)acc;
}

// Pack W2/W3 (f32 [k][n]) -> bf16 transposed+padded images [n][KPAD];
// W4 -> 16xKPAD image with row 0 = W4, rest zero.
__global__ __launch_bounds__(256) void pack_weights_kernel(
    const float* __restrict__ W2, const float* __restrict__ W3,
    const float* __restrict__ W4,
    __bf16* __restrict__ W2Tg, __bf16* __restrict__ W3Tg,
    __bf16* __restrict__ W4Ag)
{
  const int idx = blockIdx.x * 256 + threadIdx.x;
  if (idx < 128 * 128) {
    const int k = idx >> 7, n = idx & 127;
    W2Tg[n * KPAD + k] = (__bf16)W2[idx];
    W3Tg[n * KPAD + k] = (__bf16)W3[idx];
  }
  if (idx < W4ELEMS)
    W4Ag[idx] = (idx < 128) ? (__bf16)W4[idx] : (__bf16)0.0f;
}

// ---------------------------------------------------------------------------
// Main fused edge-MLP kernel. One wave = 32 edges end-to-end.
// ---------------------------------------------------------------------------
__global__ __launch_bounds__(128) void edge_mlp_kernel(
    const __bf16* __restrict__ P,
    const float*  __restrict__ graph,
    const int*    __restrict__ edge_index,
    const __bf16* __restrict__ W2Tg, const __bf16* __restrict__ W3Tg,
    const __bf16* __restrict__ W4Ag,
    const float* __restrict__ W1,    // row 256 = g-column weights
    const float* __restrict__ b2, const float* __restrict__ b3,
    const float* __restrict__ b4,
    float* __restrict__ out,
    int E, int n_nodes, int n_tiles, int n_wg)
{
  extern __shared__ char smem[];
  __bf16* W2T  = (__bf16*)smem;                 // LDS offset 0
  __bf16* W3T  = W2T + 128 * KPAD;              // LDS offset WBYTES
  __bf16* W4A  = W3T + 128 * KPAD;              // LDS offset 2*WBYTES
  float*  w1g  = (float*)(W4A + W4ELEMS);       // 128 f32
  float*  b2s  = w1g + 128;
  float*  b3s  = b2s + 128;
  __bf16* hall = (__bf16*)(b3s + 128);          // per wave: 2 * TILE_M * KPAD

  const int tid    = threadIdx.x;
  const int wave   = tid >> 5;
  const int lane   = tid & 31;
  const int lane16 = lane & 15;
  const int hihalf = lane >> 4;
  __bf16* hA = hall + wave * (2 * TILE_M * KPAD);
  __bf16* hB = hA + TILE_M * KPAD;

  // ---- stage weight images via TDM (wave 0), vectors via plain loads ----
  if (tid < 32) {
    tdm_load_to_lds(W2Tg, 0u,           128 * KPAD);
    tdm_load_to_lds(W3Tg, WBYTES,       128 * KPAD);
    tdm_load_to_lds(W4Ag, 2u * WBYTES,  W4ELEMS);
  }
  if (tid < 128) {
    w1g[tid] = W1[256 * 128 + tid];
    b2s[tid] = b2[tid];
    b3s[tid] = b3[tid];
  }
  if (tid < 32) __builtin_amdgcn_s_wait_tensorcnt(0);
  __syncthreads();

  const float bias4 = b4[0];

  // loop-invariant: W4-image A fragments + w1g slice
  v16bf a4[4];
#pragma unroll
  for (int kk = 0; kk < 4; ++kk) {
    const __bf16* wb = W4A + lane16 * KPAD + kk * 32 + hihalf * 8;
    a4[kk] = cat8(*(const v8bf*)wb, *(const v8bf*)(wb + 16));
  }
  const v4f wvf = *(const v4f*)(w1g + lane * 4);

  for (int t = blockIdx.x; t < n_tiles; t += n_wg) {
    const int baseEdge = (t * WAVES_PER_WG + wave) * TILE_M;
    const int e  = baseEdge + lane;                 // one edge per lane
    const int ec = (e < E) ? e : (E - 1);
    const int src = edge_index[ec];
    const int dst = edge_index[E + ec];
    const float g = graph[(long long)src * n_nodes + dst];

    // prefetch next tile's edge indices
    const int tn = t + n_wg;
    if (tn < n_tiles) {
      const int nb = (tn * WAVES_PER_WG + wave) * TILE_M + lane;
      __builtin_prefetch(edge_index + nb, 0, 1);
      __builtin_prefetch(edge_index + E + nb, 0, 1);
    }

    // ---------------- layer 1 (bf16 gathers, f32 math) ----------------
#pragma unroll 4
    for (int m = 0; m < TILE_M; ++m) {
      const int   sm = __shfl(src, m, 32);
      const int   dm = __shfl(dst, m, 32);
      const float gm = __shfl(g,   m, 32);
      const v4bf pa = *(const v4bf*)(P + (long long)sm * 256 + lane * 4);
      const v4bf pb = *(const v4bf*)(P + (long long)dm * 256 + 128 + lane * 4);
      v4f r = __builtin_convertvector(pa, v4f) + __builtin_convertvector(pb, v4f);
#pragma unroll
      for (int c = 0; c < 4; ++c) {
        r[c] = fmaf(gm, wvf[c], r[c]);
        r[c] = __builtin_amdgcn_fmed3f(r[c], 0.0f, __builtin_inff());
      }
      *(v4bf*)(hA + m * KPAD + lane * 4) = __builtin_convertvector(r, v4bf);
    }
    lds_wave_sync();

    // ---------------- layers 2 & 3 via WMMA (D = W^T x h^T) ----------------
#pragma unroll
    for (int layer = 0; layer < 2; ++layer) {
      const __bf16* WT  = (layer == 0) ? W2T : W3T;
      const float*  bs  = (layer == 0) ? b2s : b3s;
      const __bf16* hin = (layer == 0) ? hA  : hB;
      __bf16*       hout= (layer == 0) ? hB  : hA;

      v16bf bfr[2][4];
#pragma unroll
      for (int tle = 0; tle < 2; ++tle)
#pragma unroll
        for (int kk = 0; kk < 4; ++kk) {
          const __bf16* hb = hin + (tle * 16 + lane16) * KPAD + kk * 32 + hihalf * 8;
          bfr[tle][kk] = cat8(*(const v8bf*)hb, *(const v8bf*)(hb + 16));
        }

#pragma unroll
      for (int nt = 0; nt < 8; ++nt) {
        const float4 bl = *(const float4*)(bs + nt * 16 + hihalf * 8);
        const float4 bh = *(const float4*)(bs + nt * 16 + hihalf * 8 + 4);
        v8f acc0 = {bl.x, bl.y, bl.z, bl.w, bh.x, bh.y, bh.z, bh.w};
        v8f acc1 = acc0;
#pragma unroll
        for (int kk = 0; kk < 4; ++kk) {
          const __bf16* wb = WT + (nt * 16 + lane16) * KPAD + kk * 32 + hihalf * 8;
          const v16bf af = cat8(*(const v8bf*)wb, *(const v8bf*)(wb + 16));
          acc0 = __builtin_amdgcn_wmma_f32_16x16x32_bf16(
              false, af, false, bfr[0][kk], (short)0, acc0, false, false);
          acc1 = __builtin_amdgcn_wmma_f32_16x16x32_bf16(
              false, af, false, bfr[1][kk], (short)0, acc1, false, false);
        }
        *(v8bf*)(hout + lane16 * KPAD        + nt * 16 + hihalf * 8) = relu_cvt8(acc0);
        *(v8bf*)(hout + (16 + lane16) * KPAD + nt * 16 + hihalf * 8) = relu_cvt8(acc1);
      }
      lds_wave_sync();
    }

    // ---------------- layer 4 via WMMA: val = W4img x h3^T ----------------
    v16bf b3f[2][4];
#pragma unroll
    for (int tle = 0; tle < 2; ++tle)
#pragma unroll
      for (int kk = 0; kk < 4; ++kk) {
        const __bf16* hb = hA + (tle * 16 + lane16) * KPAD + kk * 32 + hihalf * 8;
        b3f[tle][kk] = cat8(*(const v8bf*)hb, *(const v8bf*)(hb + 16));
      }
    v8f acc0 = {0.f, 0.f, 0.f, 0.f, 0.f, 0.f, 0.f, 0.f};
    v8f acc1 = acc0;
#pragma unroll
    for (int kk = 0; kk < 4; ++kk) {
      acc0 = __builtin_amdgcn_wmma_f32_16x16x32_bf16(
          false, a4[kk], false, b3f[0][kk], (short)0, acc0, false, false);
      acc1 = __builtin_amdgcn_wmma_f32_16x16x32_bf16(
          false, a4[kk], false, b3f[1][kk], (short)0, acc1, false, false);
    }
    // D row 0 (M=0): value for edge N sits in acc[0] of lanes 0..15.
    const float s1  = __shfl(acc1[0], lane16, 32);
    const float val = (hihalf ? s1 : acc0[0]) + bias4;
    if (e < E)
      out[e] = 1.0f / (1.0f + __expf(-val));
    lds_wave_sync();
  }
}

// ---------------------------------------------------------------------------
extern "C" void kernel_launch(void* const* d_in, const int* in_sizes, int n_in,
                              void* d_out, int out_size, void* d_ws, size_t ws_size,
                              hipStream_t stream) {
  const float* z          = (const float*)d_in[0];
  const float* graph      = (const float*)d_in[2];
  const int*   edge_index = (const int*)d_in[3];
  const float* W1 = (const float*)d_in[4];
  const float* b1 = (const float*)d_in[5];
  const float* W2 = (const float*)d_in[6];
  const float* b2 = (const float*)d_in[7];
  const float* W3 = (const float*)d_in[8];
  const float* b3 = (const float*)d_in[9];
  const float* W4 = (const float*)d_in[10];
  const float* b4 = (const float*)d_in[11];
  float* out = (float*)d_out;

  const int n_nodes = in_sizes[0] / 128;
  const int E       = in_sizes[3] / 2;

  __bf16* P    = (__bf16*)d_ws;                      // n_nodes*256 bf16
  char*   wrk  = (char*)d_ws + (size_t)n_nodes * 256 * 2;
  __bf16* W2Tg = (__bf16*)wrk;                       // packed bf16 weight images
  __bf16* W3Tg = W2Tg + 128 * KPAD;
  __bf16* W4Ag = W2Tg + 2 * 128 * KPAD;

  precompute_P_kernel<<<n_nodes, 256, 0, stream>>>(z, W1, b1, P, n_nodes);
  pack_weights_kernel<<<(128 * 128 + 255) / 256, 256, 0, stream>>>(
      W2, W3, W4, W2Tg, W3Tg, W4Ag);

  const int edges_per_wg_iter = WAVES_PER_WG * TILE_M;   // 128
  const int n_tiles = (E + edges_per_wg_iter - 1) / edges_per_wg_iter;
  const int n_wg    = (n_tiles < 500) ? n_tiles : 500;

  const size_t smem =
      (size_t)2 * WBYTES + (size_t)W4ELEMS * 2        // weight images (bf16)
      + 3 * 128 * 4                                   // w1g, b2, b3 (f32)
      + (size_t)WAVES_PER_WG * 2 * TILE_M * KPAD * 2; // h double-buffers

  edge_mlp_kernel<<<n_wg, 128, smem, stream>>>(
      P, graph, edge_index, W2Tg, W3Tg, W4Ag, W1, b2, b3, b4, out,
      E, n_nodes, n_tiles, n_wg);
}
